// MultiHeadAttention_ca_7438883356886
// MI455X (gfx1250) — matrix operand order repaired and checked by reading, compiled, not run on hardware
//
#include <hip/hip_runtime.h>

typedef __bf16 bf16_t;
typedef __attribute__((ext_vector_type(16))) __bf16 v16bf;
typedef __attribute__((ext_vector_type(8)))  __bf16 v8bf;
typedef __attribute__((ext_vector_type(8)))  float  v8f;

__device__ __forceinline__ __bf16 to_bf16(float x)  { return (__bf16)x; }
__device__ __forceinline__ __bf16 to_bf16(__bf16 x) { return x; }

// A-fragment per ISA 16-bit A layout: lane<16 -> K {0..7,16..23}, lane>=16 -> +8
__device__ __forceinline__ v16bf frag_a(const __bf16* row, int ks, int khalf) {
    v8bf lo = *(const v8bf*)(row + ks + khalf * 8);
    v8bf hi = *(const v8bf*)(row + ks + 16 + khalf * 8);
    v16bf r;
    #pragma unroll
    for (int i = 0; i < 8; ++i) { r[i] = lo[i]; r[i + 8] = hi[i]; }
    return r;
}
// B-fragment matching A layout: lane<16 -> K {0..7,16..23}, lane>=16 -> +8
__device__ __forceinline__ v16bf frag_b(const __bf16* row, int ks, int khalf) {
    v8bf lo = *(const v8bf*)(row + ks + khalf * 8);
    v8bf hi = *(const v8bf*)(row + ks + 16 + khalf * 8);
    v16bf r;
    #pragma unroll
    for (int i = 0; i < 8; ++i) { r[i] = lo[i]; r[i + 8] = hi[i]; }
    return r;
}

// Async global->LDS 16B copy; inst offset applies to BOTH lds and global addr.
__device__ __forceinline__ void async_copy_b128(unsigned lds_addr, const void* gptr) {
    asm volatile("global_load_async_to_lds_b128 %0, %1, off"
                 :: "v"(lds_addr), "v"(gptr) : "memory");
}
__device__ __forceinline__ unsigned lds_off(const void* p) {
    return (unsigned)(size_t)p;   // low 32 bits of flat LDS-aperture addr == LDS_ADDR
}

// ---------------------------------------------------------------------------
// WMMA GEMM: D[M,N] = A[M,K] * B + bias
//   BT=true : B stored N-major ([N,K] row-major: element (k,n) = B[n*ldb+k])
//   BT=false: B stored K-major ([K,N] row-major)
//   OBF     : bf16 output, else fp32.
//   AASYNC  : stage A via GLOBAL_LOAD_ASYNC_TO_LDS (requires TA==bf16)
//   BASYNC  : stage B via async copy (requires TB==bf16 && BT)
// Tile: BM=128, BN=64, BK=64. 256 threads = 8 waves; each wave computes a
// 32x32 macro-tile (2x2 accumulators) -> 8 WMMAs per barrier pair.
// ---------------------------------------------------------------------------
template<typename TA, typename TB, bool BT, bool OBF, bool AASYNC, bool BASYNC>
__global__ __launch_bounds__(256)
void gemm_wmma_kernel(const TA* __restrict__ A, const TB* __restrict__ B,
                      const float* __restrict__ bias,
                      float* __restrict__ Cf, bf16_t* __restrict__ Cb,
                      int Kdim, int lda, int ldb, int ldc,
                      long sA, long sB, long sC)
{
    constexpr int BM = 128, BN = 64, BK = 64, LDK = 72; // 144B rows (16B mult)
    __shared__ alignas(16) __bf16 As[BM][LDK];
    __shared__ alignas(16) __bf16 Bt[BN][LDK];          // [n][k]

    const int tid   = threadIdx.x;
    const int wave  = tid >> 5;
    const int lane  = tid & 31;
    const int wm    = wave & 3;      // 0..3 -> 32-row macro tile
    const int wn    = wave >> 2;     // 0..1 -> 32-col macro tile
    const int lrow  = lane & 15;
    const int khalf = lane >> 4;

    const long batch = blockIdx.z;
    const TA* Ab = A + batch * sA;
    const TB* Bb = B + batch * sB;
    const int m0 = blockIdx.y * BM;
    const int n0 = blockIdx.x * BN;

    // staging assignments
    const int ar = tid >> 1,  ac = (tid & 1) * 32;   // A: 32 elems/thread (64B bf16)
    const int bn_ = tid >> 2, bk_ = (tid & 3) * 16;  // B: 16 elems/thread (32B bf16)

    v8f acc[2][2];
    #pragma unroll
    for (int i = 0; i < 2; ++i)
        #pragma unroll
        for (int j = 0; j < 2; ++j)
            #pragma unroll
            for (int e = 0; e < 8; ++e) acc[i][j][e] = 0.0f;

    for (int kb = 0; kb < Kdim; kb += BK) {
        // ---- stage A (BM x BK) ----
        const TA* aptr = Ab + (long)(m0 + ar) * lda + kb + ac;
        if (AASYNC) {
            const unsigned ld = lds_off(&As[ar][ac]);
            const char* gp = (const char*)aptr;
            #pragma unroll
            for (int j = 0; j < 4; ++j)           // 4 x 16B = 64B per thread
                async_copy_b128(ld + j * 16, gp + j * 16);
        } else {
            #pragma unroll
            for (int j = 0; j < 32; ++j) As[ar][ac + j] = to_bf16(aptr[j]);
            if (kb + BK < Kdim) __builtin_prefetch(aptr + BK, 0, 1);
        }

        // ---- stage B into Bt[n][k] ----
        if (BASYNC) {
            const char* gp = (const char*)(Bb + (long)(n0 + bn_) * ldb + kb + bk_);
            const unsigned ld = lds_off(&Bt[bn_][bk_]);
            async_copy_b128(ld, gp);              // 2 x 16B = 32B per thread
            async_copy_b128(ld + 16, gp + 16);
        } else if (BT) {
            const TB* bptr = Bb + (long)(n0 + bn_) * ldb + kb + bk_;
            #pragma unroll
            for (int j = 0; j < 16; ++j) Bt[bn_][bk_ + j] = to_bf16(bptr[j]);
        } else {
            const TB* bptr = Bb + (long)(kb + bn_) * ldb + n0 + bk_;
            #pragma unroll
            for (int j = 0; j < 16; ++j) Bt[bk_ + j][bn_] = to_bf16(bptr[j]);
        }
        if (AASYNC || BASYNC)
            asm volatile("s_wait_asynccnt 0x0" ::: "memory");
        __syncthreads();

        // ---- 2 K-steps x 2x2 tiles = 8 WMMAs per barrier pair ----
        const int arow = wm * 32 + lrow;
        const int bcol = wn * 32 + lrow;
        #pragma unroll
        for (int ks = 0; ks < BK; ks += 32) {
            v16bf a0 = frag_a(&As[arow][0],      ks, khalf);
            v16bf a1 = frag_a(&As[arow + 16][0], ks, khalf);
            v16bf b0 = frag_b(&Bt[bcol][0],      ks, khalf);
            v16bf b1 = frag_b(&Bt[bcol + 16][0], ks, khalf);
            acc[0][0] = __builtin_amdgcn_wmma_f32_16x16x32_bf16(
                            false, a0, false, b0, (short)0, acc[0][0], false, false);
            acc[0][1] = __builtin_amdgcn_wmma_f32_16x16x32_bf16(
                            false, a0, false, b1, (short)0, acc[0][1], false, false);
            acc[1][0] = __builtin_amdgcn_wmma_f32_16x16x32_bf16(
                            false, a1, false, b0, (short)0, acc[1][0], false, false);
            acc[1][1] = __builtin_amdgcn_wmma_f32_16x16x32_bf16(
                            false, a1, false, b1, (short)0, acc[1][1], false, false);
        }
        __syncthreads();
    }

    // ---- epilogue: C layout -> lane%16 = N col, VGPR r + 8*khalf = M row ----
    #pragma unroll
    for (int j = 0; j < 2; ++j) {
        const int gn = n0 + wn * 32 + j * 16 + lrow;
        const float bval = bias ? bias[gn] : 0.0f;
        #pragma unroll
        for (int i = 0; i < 2; ++i) {
            #pragma unroll
            for (int r = 0; r < 8; ++r) {
                const int gm = m0 + wm * 32 + i * 16 + khalf * 8 + r;
                const float v = acc[i][j][r] + bval;
                const long idx = batch * sC + (long)gm * ldc + gn;
                if (OBF) Cb[idx] = (__bf16)v; else Cf[idx] = v;
            }
        }
    }
}

// ---------------------------------------------------------------------------
// Row softmax over C=512 (in-place fp32) + bf16 copy for the next GEMM.
// ---------------------------------------------------------------------------
__global__ __launch_bounds__(256)
void softmax512_kernel(float* __restrict__ attn, bf16_t* __restrict__ att_bf,
                       float scale)
{
    __shared__ float red[256];
    const long row = blockIdx.x;
    float* p = attn + row * 512;
    const int t = threadIdx.x;

    float x0 = p[t], x1 = p[t + 256];
    red[t] = fmaxf(x0, x1);
    __syncthreads();
    for (int s = 128; s > 0; s >>= 1) {
        if (t < s) red[t] = fmaxf(red[t], red[t + s]);
        __syncthreads();
    }
    const float m = red[0];
    __syncthreads();

    const float e0 = __expf((x0 - m) * scale);
    const float e1 = __expf((x1 - m) * scale);
    red[t] = e0 + e1;
    __syncthreads();
    for (int s = 128; s > 0; s >>= 1) {
        if (t < s) red[t] += red[t + s];
        __syncthreads();
    }
    const float inv = 1.0f / red[0];

    const float y0 = e0 * inv, y1 = e1 * inv;
    p[t] = y0;
    p[t + 256] = y1;
    att_bf[row * 512 + t]       = (__bf16)y0;
    att_bf[row * 512 + t + 256] = (__bf16)y1;
}

// ---------------------------------------------------------------------------
// BatchNorm stats: one block per channel c, reduce over (B=128, S=128).
// ---------------------------------------------------------------------------
__global__ __launch_bounds__(256)
void bn_stats_kernel(const float* __restrict__ pre,
                     float* __restrict__ mean, float* __restrict__ invstd)
{
    constexpr int B = 128, C = 512, S = 128, N = B * S;
    __shared__ float ssum[256], ssq[256];
    const int c = blockIdx.x, t = threadIdx.x;
    float sum = 0.0f, sq = 0.0f;
    for (int i = t; i < N; i += 256) {
        const int b = i / S, s = i - b * S;
        const float v = pre[((long)b * C + c) * S + s];
        sum += v; sq += v * v;
    }
    ssum[t] = sum; ssq[t] = sq;
    __syncthreads();
    for (int st = 128; st > 0; st >>= 1) {
        if (t < st) { ssum[t] += ssum[t + st]; ssq[t] += ssq[t + st]; }
        __syncthreads();
    }
    if (t == 0) {
        const float mu  = ssum[0] / (float)N;
        const float var = ssq[0] / (float)N - mu * mu;
        mean[c]   = mu;
        invstd[c] = rsqrtf(var + 1e-5f);
    }
}

// ---------------------------------------------------------------------------
// out = value + relu((pre - mean[c])*invstd[c]*gamma[c] + beta[c])
// ---------------------------------------------------------------------------
__global__ __launch_bounds__(256)
void bn_finalize_kernel(const float* __restrict__ pre, const float* __restrict__ value,
                        const float* __restrict__ mean, const float* __restrict__ invstd,
                        const float* __restrict__ gamma, const float* __restrict__ beta,
                        float* __restrict__ out, long total)
{
    constexpr int C = 512, S = 128;
    const long i = (long)blockIdx.x * 256 + threadIdx.x;
    if (i >= total) return;
    const int c = (int)((i / S) % C);
    float x = (pre[i] - mean[c]) * invstd[c] * gamma[c] + beta[c];
    x = fmaxf(x, 0.0f);
    out[i] = value[i] + x;
}

// ---------------------------------------------------------------------------
extern "C" void kernel_launch(void* const* d_in, const int* in_sizes, int n_in,
                              void* d_out, int out_size, void* d_ws, size_t ws_size,
                              hipStream_t stream)
{
    (void)in_sizes; (void)n_in; (void)out_size; (void)ws_size;

    const float* query = (const float*)d_in[0];
    const float* key_  = (const float*)d_in[1];
    const float* value = (const float*)d_in[2];
    const float* Wq = (const float*)d_in[3];
    const float* bq = (const float*)d_in[4];
    const float* Wk = (const float*)d_in[5];
    const float* bk = (const float*)d_in[6];
    const float* Wv = (const float*)d_in[7];
    const float* bv = (const float*)d_in[8];
    const float* Wf = (const float*)d_in[9];
    const float* bf = (const float*)d_in[10];
    const float* gamma = (const float*)d_in[11];
    const float* beta  = (const float*)d_in[12];

    constexpr int B = 128, C = 512, S = 128, M = 512;
    constexpr long BCM = (long)B * C * M;
    constexpr long BCC = (long)B * C * C;
    constexpr long BCS = (long)B * C * S;

    // ---- workspace layout (bf16 staging + stats), with buffer reuse ----
    char* ws = (char*)d_ws;
    bf16_t* qb = (bf16_t*)ws;       ws += BCM * 2;
    bf16_t* kb = (bf16_t*)ws;       ws += BCM * 2;
    bf16_t* vb = (bf16_t*)ws;       ws += BCM * 2;
    bf16_t* att_bf = (bf16_t*)ws;   ws += BCC * 2;
    float* mean   = (float*)ws;     ws += C * 4;
    float* invstd = (float*)ws;     ws += C * 4;
    bf16_t* ctx = qb;               // q consumed after scores GEMM
    float*  pre = (float*)vb;       // v consumed after context GEMM

    float* out0 = (float*)d_out;    // [B,C,S]
    float* attn = out0 + BCS;       // [B,C,C] second output; also scores scratch

    const dim3 blk(256);

    // ---- Q/K/V projections: [B*C,S] x W[M,S]^T + b -> bf16 [B*C,M] ----
    const dim3 gproj(M / 64, (B * C) / 128, 1);
    gemm_wmma_kernel<float, float, true, true, false, false><<<gproj, blk, 0, stream>>>(
        query, Wq, bq, nullptr, qb, S, S, S, M, 0, 0, 0);
    gemm_wmma_kernel<float, float, true, true, false, false><<<gproj, blk, 0, stream>>>(
        key_, Wk, bk, nullptr, kb, S, S, S, M, 0, 0, 0);
    gemm_wmma_kernel<float, float, true, true, false, false><<<gproj, blk, 0, stream>>>(
        value, Wv, bv, nullptr, vb, S, S, S, M, 0, 0, 0);

    // ---- scores: per-batch q[C,M] x k[C,M]^T -> fp32 into d_out attn region ----
    const dim3 gsc(C / 64, C / 128, B);
    gemm_wmma_kernel<bf16_t, bf16_t, true, false, true, true><<<gsc, blk, 0, stream>>>(
        qb, kb, nullptr, attn, nullptr, M, M, M, C,
        (long)C * M, (long)C * M, (long)C * C);

    // ---- softmax (in-place fp32) + bf16 copy; scale folded in ----
    const float scale = 0.044194173824159216f;   // 512^-0.5
    softmax512_kernel<<<dim3((unsigned)(B * C)), blk, 0, stream>>>(attn, att_bf, scale);

    // ---- context: att[C,C] x v[C,M] -> bf16 ctx ----
    const dim3 gctx(M / 64, C / 128, B);
    gemm_wmma_kernel<bf16_t, bf16_t, false, true, true, false><<<gctx, blk, 0, stream>>>(
        att_bf, vb, nullptr, nullptr, ctx, C, C, M, M,
        (long)C * C, (long)C * M, (long)C * M);

    // ---- output proj: ctx[B*C,M] x Wf[S,M]^T + bf -> fp32 pre-BN ----
    const dim3 gout(S / 64, (B * C) / 128, 1);
    gemm_wmma_kernel<bf16_t, float, true, false, true, false><<<gout, blk, 0, stream>>>(
        ctx, Wf, bf, pre, nullptr, M, M, M, S, 0, 0, 0);

    // ---- BatchNorm (training stats) + ReLU + residual ----
    bn_stats_kernel<<<dim3(C), blk, 0, stream>>>(pre, mean, invstd);
    bn_finalize_kernel<<<dim3((unsigned)(BCS / 256)), blk, 0, stream>>>(
        pre, value, mean, invstd, gamma, beta, out0, BCS);
}
